// ContrastiveLoss_57801669869809
// MI455X (gfx1250) — compile-verified, matrix operand order checked by reading, not run on hardware
//
#include <hip/hip_runtime.h>
#include <hip/hip_bf16.h>
#include <stdint.h>

typedef __attribute__((ext_vector_type(16))) __bf16 v16bf;
typedef __attribute__((ext_vector_type(8)))  float  v8f;
typedef __attribute__((ext_vector_type(4)))  int    v4i;

#define TM 128      // workgroup M tile
#define TN 256      // workgroup N tile
#define KC 64       // bf16 K elements staged per LDS round (2 WMMA k-steps)
#define LDSP 72     // padded LDS row stride (bf16 elems, 16B aligned, bank-conflict free)
#define MARGIN_F 2.0f

#if __has_builtin(__builtin_amdgcn_global_load_async_to_lds_b128)
#define ASYNC_LDS 1
typedef __attribute__((address_space(1))) v4i as1_v4i;   // global int4
typedef __attribute__((address_space(3))) v4i as3_v4i;   // LDS int4
#else
#define ASYNC_LDS 0
#endif

// ------------------------------------------------------------------
// Kernel 1: row norms, per-row positive loss, fp32 -> bf16 conversion
// ------------------------------------------------------------------
__global__ __launch_bounds__(256) void norms_pos_cvt(
    const float* __restrict__ o1, const float* __restrict__ o2,
    float* __restrict__ n1, float* __restrict__ n2,
    unsigned short* __restrict__ a16, unsigned short* __restrict__ b16,
    float* __restrict__ posr, int D)
{
  const int row = blockIdx.x;
  const size_t base = (size_t)row * (size_t)D;
  float s1 = 0.f, s2 = 0.f, p = 0.f;
  for (int d = threadIdx.x; d < D; d += 256) {
    float a = o1[base + d];
    float b = o2[base + d];
    s1 = fmaf(a, a, s1);
    s2 = fmaf(b, b, s2);
    float df = b - a;
    p = fmaf(df, df, p);
    // round-to-nearest-even bf16
    uint32_t ua = __float_as_uint(a);
    uint32_t ub = __float_as_uint(b);
    ua += 0x7FFFu + ((ua >> 16) & 1u);
    ub += 0x7FFFu + ((ub >> 16) & 1u);
    a16[base + d] = (unsigned short)(ua >> 16);
    b16[base + d] = (unsigned short)(ub >> 16);
  }
  __shared__ float red[3][256];
  red[0][threadIdx.x] = s1; red[1][threadIdx.x] = s2; red[2][threadIdx.x] = p;
  __syncthreads();
  for (int off = 128; off > 0; off >>= 1) {
    if ((int)threadIdx.x < off) {
      red[0][threadIdx.x] += red[0][threadIdx.x + off];
      red[1][threadIdx.x] += red[1][threadIdx.x + off];
      red[2][threadIdx.x] += red[2][threadIdx.x + off];
    }
    __syncthreads();
  }
  if (threadIdx.x == 0) {
    n1[row] = red[0][0];
    n2[row] = red[1][0];
    posr[row] = red[2][0];
  }
}

// ------------------------------------------------------------------
// Kernel 2: pairwise distance GEMM chunk via v_wmma_f32_16x16x32_bf16
//   Block tile 128x256, 8 waves of 64x64 (4x4 WMMA tiles each)
//   sq[r - rowBase][n] = max(0, n1[r] + n2[n] - 2 * <o1_r, o2_n>)
// ------------------------------------------------------------------
__global__ __launch_bounds__(256) void pair_gemm(
    const unsigned short* __restrict__ A16, const unsigned short* __restrict__ B16,
    const float* __restrict__ n1, const float* __restrict__ n2,
    float* __restrict__ sqout, int rowBase, int D, int Ncols)
{
  __shared__ __align__(16) unsigned short As[TM * LDSP];
  __shared__ __align__(16) unsigned short Bs[TN * LDSP];

  const int tid  = threadIdx.x;
  const int lane = tid & 31;
  const int wid  = tid >> 5;          // 8 waves: 2 (M) x 4 (N)
  const int mw   = (wid & 1) * 64;    // wave M offset
  const int nw   = (wid >> 1) * 64;   // wave N offset
  const int hl   = lane >> 4;         // half-wave selector (K chunk)
  const int ml   = lane & 15;         // M / N index within 16-tile

  const int mBlock = rowBase + blockIdx.y * TM;
  const int nBlock = blockIdx.x * TN;

  v8f acc[4][4];
#pragma unroll
  for (int i = 0; i < 4; i++)
#pragma unroll
    for (int j = 0; j < 4; j++)
#pragma unroll
      for (int v = 0; v < 8; v++) acc[i][j][v] = 0.0f;

  for (int kt = 0; kt < D; kt += KC) {
    __syncthreads();
    // ---- stage A: 128 rows x 64 bf16, 1024 x 16B segments ----
#pragma unroll
    for (int it = 0; it < (TM * KC / 8) / 256; it++) {
      int s   = tid + it * 256;
      int r   = s >> 3;               // row 0..127
      int seg = s & 7;                // 8 x 16B segments per row
      const unsigned short* g = A16 + (size_t)(mBlock + r) * D + kt + seg * 8;
      unsigned short* l = &As[r * LDSP + seg * 8];
#if ASYNC_LDS
      __builtin_amdgcn_global_load_async_to_lds_b128((as1_v4i*)g, (as3_v4i*)l, 0, 0);
#else
      *(uint4*)l = *(const uint4*)g;
#endif
      if (kt + KC < D)
        __builtin_prefetch(g + KC, 0, 0);
    }
    // ---- stage B: 256 rows x 64 bf16, 2048 x 16B segments ----
#pragma unroll
    for (int it = 0; it < (TN * KC / 8) / 256; it++) {
      int s   = tid + it * 256;
      int r   = s >> 3;               // row 0..255
      int seg = s & 7;
      const unsigned short* g = B16 + (size_t)(nBlock + r) * D + kt + seg * 8;
      unsigned short* l = &Bs[r * LDSP + seg * 8];
#if ASYNC_LDS
      __builtin_amdgcn_global_load_async_to_lds_b128((as1_v4i*)g, (as3_v4i*)l, 0, 0);
#else
      *(uint4*)l = *(const uint4*)g;
#endif
      if (kt + KC < D)
        __builtin_prefetch(g + KC, 0, 0);
    }
#if ASYNC_LDS
    asm volatile("s_wait_asynccnt 0x0" ::: "memory");
#endif
    __syncthreads();

#pragma unroll
    for (int kk = 0; kk < KC; kk += 32) {
      union Frag { uint4 q[2]; v16bf v; };
      Frag a[4], b[4];
      // A fragment: lane holds K = kk+8*hl..+7 (V0..3) and kk+16+8*hl..+7 (V4..7)
#pragma unroll
      for (int t = 0; t < 4; t++) {
        const unsigned short* p = &As[(mw + t * 16 + ml) * LDSP + kk + hl * 8];
        a[t].q[0] = *(const uint4*)(p);
        a[t].q[1] = *(const uint4*)(p + 16);
      }
#pragma unroll
      for (int t = 0; t < 4; t++) {
        const unsigned short* p = &Bs[(nw + t * 16 + ml) * LDSP + kk + hl * 8];
        b[t].q[0] = *(const uint4*)(p);
        b[t].q[1] = *(const uint4*)(p + 16);
      }
#pragma unroll
      for (int i = 0; i < 4; i++)
#pragma unroll
        for (int j = 0; j < 4; j++)
          acc[i][j] = __builtin_amdgcn_wmma_f32_16x16x32_bf16(
              false, a[i].v, false, b[j].v, (short)0, acc[i][j], false, false);
    }
  }

  // epilogue: sq = clamp(n1 + n2 - 2*dot, 0)
  float n1c[4][8];
#pragma unroll
  for (int i = 0; i < 4; i++)
#pragma unroll
    for (int v = 0; v < 8; v++)
      n1c[i][v] = n1[rowBase + blockIdx.y * TM + mw + i * 16 + hl * 8 + v];

#pragma unroll
  for (int i = 0; i < 4; i++) {
#pragma unroll
    for (int j = 0; j < 4; j++) {
      int n = nBlock + nw + j * 16 + ml;
      float nn2 = n2[n];
#pragma unroll
      for (int v = 0; v < 8; v++) {
        int mrow = blockIdx.y * TM + mw + i * 16 + hl * 8 + v;   // row within chunk
        float s = n1c[i][v] + nn2 - 2.0f * acc[i][j][v];
        s = fmaxf(s, 0.0f);
        sqout[(size_t)mrow * Ncols + n] = s;
      }
    }
  }
}

// ------------------------------------------------------------------
// Kernel 3: per-row rank selection (byte radix-select over 8192 keys)
// ------------------------------------------------------------------
__global__ __launch_bounds__(256) void select_rank(
    const float* __restrict__ sq, const int* __restrict__ rn,
    const int* __restrict__ quantp,
    float* __restrict__ negr, int rowBase, int Ncols)
{
  __shared__ uint32_t keys[8192];     // Ncols == 8192
  __shared__ uint32_t hist[256];
  __shared__ uint32_t scal[4];        // 0: cntLess, 1: cntEqBefore, 2: bin, 3: cumExcl

  const int tid = threadIdx.x;
  const int row = rowBase + blockIdx.x;   // global row == self column index
  const float* srow = sq + (size_t)blockIdx.x * Ncols;

  // load row -> monotonic uint keys (all values >= 0 after clamp)
  for (int c = tid; c < Ncols / 4; c += 256) {
    float4 f = ((const float4*)srow)[c];
    keys[c * 4 + 0] = __float_as_uint(f.x);
    keys[c * 4 + 1] = __float_as_uint(f.y);
    keys[c * 4 + 2] = __float_as_uint(f.z);
    keys[c * 4 + 3] = __float_as_uint(f.w);
  }
  if (tid < 4) scal[tid] = 0;
  __syncthreads();

  // rank of the self-match (top_k ties break by ascending index)
  const uint32_t skey = keys[row];
  uint32_t less = 0, eq = 0;
  for (int c = tid; c < Ncols; c += 256) {
    uint32_t k = keys[c];
    less += (k < skey) ? 1u : 0u;
    eq   += (k == skey && c < row) ? 1u : 0u;
  }
  atomicAdd(&scal[0], less);
  atomicAdd(&scal[1], eq);
  __syncthreads();

  const int quant = min(quantp[0], Ncols - 1);
  const int myrn = rn[row];
  const int rank_self = (int)(scal[0] + scal[1]);
  int k = (rank_self == myrn) ? (myrn + 1) % quant : myrn;

  // 4-pass byte radix select for the k-th smallest key
  uint32_t prefix = 0;
  for (int b = 3; b >= 0; b--) {
    hist[tid] = 0;
    __syncthreads();
    const int shiftHi = (b + 1) * 8;
    for (int c = tid; c < Ncols; c += 256) {
      uint32_t kk = keys[c];
      bool match = (b == 3) || ((kk >> shiftHi) == (prefix >> shiftHi));
      if (match) atomicAdd(&hist[(kk >> (b * 8)) & 255u], 1u);
    }
    __syncthreads();
    // inclusive scan (Hillis-Steele)
    for (int off = 1; off < 256; off <<= 1) {
      uint32_t v = hist[tid];
      uint32_t add = (tid >= off) ? hist[tid - off] : 0u;
      __syncthreads();
      hist[tid] = v + add;
      __syncthreads();
    }
    uint32_t cumIncl = hist[tid];
    uint32_t cumExcl = (tid == 0) ? 0u : hist[tid - 1];
    if ((uint32_t)k >= cumExcl && (uint32_t)k < cumIncl) {
      scal[2] = (uint32_t)tid;
      scal[3] = cumExcl;
    }
    __syncthreads();
    prefix |= scal[2] << (b * 8);
    k -= (int)scal[3];
    __syncthreads();
  }

  if (tid == 0) {
    float v = __uint_as_float(prefix);
    float d = MARGIN_F - sqrtf(v);
    negr[row] = (d > 0.f) ? d : 0.f;
  }
}

// ------------------------------------------------------------------
// Kernel 4: deterministic final reduction
// ------------------------------------------------------------------
__global__ __launch_bounds__(256) void reduce_final(
    const float* __restrict__ posr, const float* __restrict__ negr,
    float* __restrict__ out, int N)
{
  __shared__ float rp[256], rq[256];
  float sp = 0.f, sn = 0.f;
  for (int i = threadIdx.x; i < N; i += 256) { sp += posr[i]; sn += negr[i]; }
  rp[threadIdx.x] = sp; rq[threadIdx.x] = sn;
  __syncthreads();
  for (int off = 128; off > 0; off >>= 1) {
    if ((int)threadIdx.x < off) {
      rp[threadIdx.x] += rp[threadIdx.x + off];
      rq[threadIdx.x] += rq[threadIdx.x + off];
    }
    __syncthreads();
  }
  if (threadIdx.x == 0) out[0] = (rp[0] + rq[0]) / (float)N;
}

// ------------------------------------------------------------------
extern "C" void kernel_launch(void* const* d_in, const int* in_sizes, int n_in,
                              void* d_out, int out_size, void* d_ws, size_t ws_size,
                              hipStream_t stream)
{
  const float* o1     = (const float*)d_in[0];
  const float* o2     = (const float*)d_in[1];
  const int*   rn     = (const int*)d_in[2];
  const int*   quantp = (const int*)d_in[3];

  const int N = in_sizes[2];            // 8192
  const int D = in_sizes[0] / N;        // 1024

  // workspace layout (all 16B aligned)
  char* ws = (char*)d_ws;
  float* posr = (float*)ws;                                   // N floats
  float* negr = posr + N;                                     // N floats
  float* n1   = negr + N;                                     // N floats
  float* n2   = n1 + N;                                       // N floats
  unsigned short* A16 = (unsigned short*)(n2 + N);            // N*D bf16
  unsigned short* B16 = A16 + (size_t)N * D;                  // N*D bf16
  float* sqbuf = (float*)(B16 + (size_t)N * D);               // R*N floats

  const size_t baseBytes = (size_t)4 * N * sizeof(float)
                         + (size_t)2 * N * D * sizeof(unsigned short);
  int R = 128;
  for (int cand = N; cand >= 128; cand >>= 1) {
    if (baseBytes + (size_t)cand * (size_t)N * sizeof(float) <= ws_size) { R = cand; break; }
  }

  norms_pos_cvt<<<N, 256, 0, stream>>>(o1, o2, n1, n2, A16, B16, posr, D);

  for (int rowBase = 0; rowBase < N; rowBase += R) {
    dim3 g(N / TN, R / TM);
    pair_gemm<<<g, 256, 0, stream>>>(A16, B16, n1, n2, sqbuf, rowBase, D, N);
    select_rank<<<R, 256, 0, stream>>>(sqbuf, rn, quantp, negr, rowBase, N);
  }

  reduce_final<<<1, 256, 0, stream>>>(posr, negr, (float*)d_out, N);
}
